// Net_7060926234635
// MI455X (gfx1250) — matrix operand order repaired and checked by reading, not compiled, not run on hardware
//
#include <hip/hip_runtime.h>
#include <hip/hip_bf16.h>

// ---------------------------------------------------------------------------
// GCN mesh-unpool pipeline for MI455X (gfx1250, wave32).
//
//   out[d] = relu( dis[d] * sum_{s->d + self} dis[s]*(x[s] @ W) + b )
//
// Memory-bound problem (~2.5 GFLOP vs hundreds of MB of edge traffic), so:
//   * GEMM uses v_wmma_f32_16x16x32_f16: one wave = one 16x16 D tile. dis[s]
//     folded into A rows (linearity), previous layer's unpool gather folded
//     into the A-row load, epilogue dual-stores g + scatter-accumulator
//     (self-loop init) -> no memsets, no gathered intermediates.
//   * Interior tiles take a FULL fast path: unguarded b128 A loads and
//     immediate-offset stores; only the one partial boundary M-tile takes
//     the guarded path. Branch is wave-uniform, so EXEC is all-ones at WMMA.
//   * Edge scatter: wave-per-edge, lane-per-feature -> coalesced 128B row
//     reads + consecutive-address global_atomic_add_f32 (L2 atomics).
//   * relu/bias in place on the accumulator; next layer reads it via unpool.
// ---------------------------------------------------------------------------

typedef __attribute__((ext_vector_type(16))) _Float16 v16h;
typedef __attribute__((ext_vector_type(8)))  float    v8f;

// ---------------- small utility kernels ----------------

__global__ void k_fill(float* __restrict__ p, float v, int n) {
  int i = blockIdx.x * blockDim.x + threadIdx.x;
  if (i < n) p[i] = v;
}

__global__ void k_deg(const int* __restrict__ dst, float* __restrict__ deg, int E) {
  int i = blockIdx.x * blockDim.x + threadIdx.x;
  if (i < E) {
    __builtin_prefetch(dst + i + 8192, 0, 1);   // stream-ahead (global_prefetch_b8)
    atomicAdd(&deg[dst[i]], 1.0f);
  }
}

__global__ void k_rsqrt(float* __restrict__ d, int n) {
  int i = blockIdx.x * blockDim.x + threadIdx.x;
  if (i < n) d[i] = rsqrtf(d[i]);              // deg >= 1 always (self-loop)
}

// ---------------- WMMA GEMM tile body ----------------
//
// Layouts per CDNA5 ISA 7.12.2 (wave32):
//   A (16x32 f16): lane l, m = l&15; half = l>>4
//       a[i]   = A[m][kc*32 + half*8 + i]        (i = 0..7)
//       a[8+i] = A[m][kc*32 + 16 + half*8 + i]
//   B (32x16 f16): lane l, n = l&15
//       b[i]   = B[kc*32 + half*8 + i][n]        (i = 0..7)
//       b[8+i] = B[kc*32 + 16 + half*8 + i][n]
//   C/D (16x16 f32): c[r] = D[m0 + r + half*8][n0 + (l&15)]

template<int FIN, int FOUT, bool FULL>
__device__ __forceinline__ void gemm_tile(
    const float* __restrict__ X, int ldx,
    const int*   __restrict__ idx,      // unpool gather (may be null)
    const float* __restrict__ dis,      // per-node of THIS layer
    const float* __restrict__ W,        // FIN x FOUT row-major
    float* __restrict__ g, float* __restrict__ acc,
    int N, int m0, int n0, int lane) {

  const int  half = lane >> 4;
  const int  l16  = lane & 15;
  const int  am   = m0 + l16;                          // A row this lane feeds
  const bool mvalid = FULL || (am < N);
  const int  arow = mvalid ? (idx ? idx[am] : am) : 0;
  const float scale = mvalid ? dis[am] : 0.0f;         // dis of THIS layer's node
  const float* xrow = X + (long)arow * ldx;
  const int  cn = n0 + l16;                            // B/D column this lane holds

  v8f c = {0.f, 0.f, 0.f, 0.f, 0.f, 0.f, 0.f, 0.f};
  constexpr int nK = (FIN + 31) / 32;

  #pragma unroll
  for (int kc = 0; kc < nK; ++kc) {
    float fa[16];
    if constexpr (FIN >= 32) {
      // row stride 32/64 floats from 256-float-aligned buffers -> b128 loads
      const float4* q = reinterpret_cast<const float4*>(xrow + kc * 32);
      float4 t0, t1, t2, t3;
      if (mvalid) {
        t0 = q[half * 2];      t1 = q[half * 2 + 1];
        t2 = q[4 + half * 2];  t3 = q[4 + half * 2 + 1];
      } else {
        t0 = t1 = t2 = t3 = float4{0.f, 0.f, 0.f, 0.f};
      }
      fa[0]=t0.x;  fa[1]=t0.y;  fa[2]=t0.z;  fa[3]=t0.w;
      fa[4]=t1.x;  fa[5]=t1.y;  fa[6]=t1.z;  fa[7]=t1.w;
      fa[8]=t2.x;  fa[9]=t2.y;  fa[10]=t2.z; fa[11]=t2.w;
      fa[12]=t3.x; fa[13]=t3.y; fa[14]=t3.z; fa[15]=t3.w;
    } else {
      #pragma unroll
      for (int i = 0; i < 8; ++i) {
        const int k0 = kc * 32 + half * 8 + i;
        const int k1 = kc * 32 + 16 + half * 8 + i;
        fa[i]     = (mvalid && k0 < FIN) ? xrow[k0] : 0.f;
        fa[8 + i] = (mvalid && k1 < FIN) ? xrow[k1] : 0.f;
      }
    }

    v16h a, b;
    #pragma unroll
    for (int i = 0; i < 16; ++i) a[i] = (_Float16)(fa[i] * scale);

    #pragma unroll
    for (int i = 0; i < 16; ++i) {
      const int k = kc * 32 + (i < 8 ? half * 8 + i : 16 + half * 8 + (i - 8));
      float w = 0.f;
      if (k < FIN && (FOUT >= 16 || cn < FOUT)) w = W[k * FOUT + cn];
      b[i] = (_Float16)w;
    }

    c = __builtin_amdgcn_wmma_f32_16x16x32_f16(
            /*neg_a=*/false, a, /*neg_b=*/false, b,
            /*c_mod=*/(short)0, c, /*reuse_a=*/false, /*reuse_b=*/false);
  }

  if (FOUT >= 16 || cn < FOUT) {
    const long base = (long)(m0 + half * 8) * FOUT + cn;
    float* gp = g   + base;
    float* ap = acc + base;
    #pragma unroll
    for (int r = 0; r < 8; ++r) {
      if (FULL || (m0 + half * 8 + r) < N) {
        gp[r * FOUT] = c[r];   // message payload (const imm offset on fast path)
        ap[r * FOUT] = c[r];   // self-loop term initializes scatter accumulator
      }
    }
  }
}

template<int FIN, int FOUT>
__global__ __launch_bounds__(256)
void k_gcn_gemm(const float* __restrict__ X, int ldx,
                const int*   __restrict__ idx,
                const float* __restrict__ dis,
                const float* __restrict__ W,
                float* __restrict__ g, float* __restrict__ acc, int N) {
  const int lane = threadIdx.x & 31;
  const int wave = threadIdx.x >> 5;
  const int wpb  = blockDim.x >> 5;
  constexpr int nt_n = (FOUT + 15) / 16;
  const int mt_n = (N + 15) / 16;
  const int tile = blockIdx.x * wpb + wave;            // wave-uniform
  if (tile >= mt_n * nt_n) return;                     // whole wave exits together
  const int m0 = (tile / nt_n) * 16;
  const int n0 = (tile % nt_n) * 16;

  if (m0 + 16 <= N)   // wave-uniform: EXEC all-ones inside either path
    gemm_tile<FIN, FOUT, true >(X, ldx, idx, dis, W, g, acc, N, m0, n0, lane);
  else
    gemm_tile<FIN, FOUT, false>(X, ldx, idx, dis, W, g, acc, N, m0, n0, lane);
}

// ---------------- edge scatter: acc[dst] += g[src] ----------------

template<int F>
__global__ void k_scatter_wave(const int* __restrict__ src, const int* __restrict__ dst,
                               const float* __restrict__ g, float* __restrict__ acc, int E) {
  const int e    = blockIdx.x * (blockDim.x >> 5) + (threadIdx.x >> 5);
  const int lane = threadIdx.x & 31;
  if (e >= E) return;
  __builtin_prefetch(src + e + 2048, 0, 1);
  const int s = src[e], d = dst[e];
  const float* gr = g   + (long)s * F;
  float*       ar = acc + (long)d * F;
  #pragma unroll
  for (int f = lane; f < F; f += 32) atomicAdd(&ar[f], gr[f]);
}

__global__ void k_scatter3(const int* __restrict__ src, const int* __restrict__ dst,
                           const float* __restrict__ g, float* __restrict__ acc, int E) {
  const int e = blockIdx.x * blockDim.x + threadIdx.x;
  if (e >= E) return;
  const int s = src[e], d = dst[e];
  atomicAdd(&acc[(long)d * 3 + 0], g[(long)s * 3 + 0]);
  atomicAdd(&acc[(long)d * 3 + 1], g[(long)s * 3 + 1]);
  atomicAdd(&acc[(long)d * 3 + 2], g[(long)s * 3 + 2]);
}

// ---------------- in-place epilogue: acc = relu(dis*acc + b) ----------------

template<int F>
__global__ void k_finalize(float* __restrict__ acc, const float* __restrict__ dis,
                           const float* __restrict__ b, long total) {
  const long t = (long)blockIdx.x * blockDim.x + threadIdx.x;
  if (t >= total) return;
  const long v = t / F;                 // F is a compile-time constant
  const int  f = (int)(t - v * F);
  acc[t] = fmaxf(fmaf(dis[v], acc[t], b[f]), 0.f);
}

// ---------------- final unpool gather to output ----------------

__global__ void k_gather3(const float* __restrict__ acc, const int* __restrict__ up,
                          float* __restrict__ out, int N4) {
  const int i = blockIdx.x * blockDim.x + threadIdx.x;
  if (i >= N4) return;
  const long u = (long)up[i] * 3;
  out[(long)i * 3 + 0] = acc[u + 0];
  out[(long)i * 3 + 1] = acc[u + 1];
  out[(long)i * 3 + 2] = acc[u + 2];
}

// ---------------------------------------------------------------------------

static inline int cdiv(long a, long b) { return (int)((a + b - 1) / b); }

extern "C" void kernel_launch(void* const* d_in, const int* in_sizes, int n_in,
                              void* d_out, int out_size, void* d_ws, size_t ws_size,
                              hipStream_t stream) {
  const float* x  = (const float*)d_in[0];
  const float* W1 = (const float*)d_in[1];  const float* b1 = (const float*)d_in[2];
  const float* W2 = (const float*)d_in[3];  const float* b2 = (const float*)d_in[4];
  const float* W3 = (const float*)d_in[5];  const float* b3 = (const float*)d_in[6];
  const float* W4 = (const float*)d_in[7];  const float* b4 = (const float*)d_in[8];
  const int* e0  = (const int*)d_in[9];
  const int* up1 = (const int*)d_in[10];
  const int* e1  = (const int*)d_in[11];
  const int* up2 = (const int*)d_in[12];
  const int* e2  = (const int*)d_in[13];
  const int* up3 = (const int*)d_in[14];
  const int* e3  = (const int*)d_in[15];
  const int* up4 = (const int*)d_in[16];

  const int N0 = in_sizes[0] / 3;
  const int E0 = in_sizes[9] / 2;
  const int N1 = in_sizes[10];
  const int E1 = in_sizes[11] / 2;
  const int N2 = in_sizes[12];
  const int E2 = in_sizes[13] / 2;
  const int N3 = in_sizes[14];
  const int E3 = in_sizes[15] / 2;
  const int N4 = in_sizes[16];

  // ---- workspace carve-up (floats, 256-float aligned => 16B row alignment) ----
  auto rup = [](size_t v) { return (v + 255) & ~(size_t)255; };
  auto mx  = [](size_t a, size_t b) { return a > b ? a : b; };
  const size_t disN = rup((size_t)N3);                                     // largest N
  const size_t gN   = rup(mx(mx((size_t)N0 * 32, (size_t)N1 * 64),
                             mx((size_t)N2 * 32, (size_t)N3 * 3)));
  const size_t aN   = rup(mx((size_t)N0 * 32, (size_t)N2 * 32));           // layers 1,3
  float* dis  = (float*)d_ws;
  float* g    = dis  + disN;
  float* accA = g    + gN;
  float* accB = accA + aN;                                                 // layers 2,4

  const int T = 256;                    // 8 waves / workgroup
  const int WPB = T / 32;

  // ---------------- layer 1: 3 -> 32, N0, identity input ----------------
  {
    const int* src = e0; const int* dst = e0 + E0;
    k_fill  <<<cdiv(N0, T), T, 0, stream>>>(dis, 1.0f, N0);
    k_deg   <<<cdiv(E0, T), T, 0, stream>>>(dst, dis, E0);
    k_rsqrt <<<cdiv(N0, T), T, 0, stream>>>(dis, N0);
    const int tiles = cdiv(N0, 16) * 2;
    k_gcn_gemm<3, 32><<<cdiv(tiles, WPB), T, 0, stream>>>(x, 3, nullptr, dis, W1, g, accA, N0);
    k_scatter_wave<32><<<cdiv(E0, WPB), T, 0, stream>>>(src, dst, g, accA, E0);
    k_finalize<32><<<cdiv((long)N0 * 32, T), T, 0, stream>>>(accA, dis, b1, (long)N0 * 32);
  }
  // ---------------- layer 2: 32 -> 64, N1, gather up1 from accA ----------------
  {
    const int* src = e1; const int* dst = e1 + E1;
    k_fill  <<<cdiv(N1, T), T, 0, stream>>>(dis, 1.0f, N1);
    k_deg   <<<cdiv(E1, T), T, 0, stream>>>(dst, dis, E1);
    k_rsqrt <<<cdiv(N1, T), T, 0, stream>>>(dis, N1);
    const int tiles = cdiv(N1, 16) * 4;
    k_gcn_gemm<32, 64><<<cdiv(tiles, WPB), T, 0, stream>>>(accA, 32, up1, dis, W2, g, accB, N1);
    k_scatter_wave<64><<<cdiv(E1, WPB), T, 0, stream>>>(src, dst, g, accB, E1);
    k_finalize<64><<<cdiv((long)N1 * 64, T), T, 0, stream>>>(accB, dis, b2, (long)N1 * 64);
  }
  // ---------------- layer 3: 64 -> 32, N2, gather up2 from accB ----------------
  {
    const int* src = e2; const int* dst = e2 + E2;
    k_fill  <<<cdiv(N2, T), T, 0, stream>>>(dis, 1.0f, N2);
    k_deg   <<<cdiv(E2, T), T, 0, stream>>>(dst, dis, E2);
    k_rsqrt <<<cdiv(N2, T), T, 0, stream>>>(dis, N2);
    const int tiles = cdiv(N2, 16) * 2;
    k_gcn_gemm<64, 32><<<cdiv(tiles, WPB), T, 0, stream>>>(accB, 64, up2, dis, W3, g, accA, N2);
    k_scatter_wave<32><<<cdiv(E2, WPB), T, 0, stream>>>(src, dst, g, accA, E2);
    k_finalize<32><<<cdiv((long)N2 * 32, T), T, 0, stream>>>(accA, dis, b3, (long)N2 * 32);
  }
  // ---------------- layer 4: 32 -> 3, N3, gather up3 from accA ----------------
  {
    const int* src = e3; const int* dst = e3 + E3;
    k_fill  <<<cdiv(N3, T), T, 0, stream>>>(dis, 1.0f, N3);
    k_deg   <<<cdiv(E3, T), T, 0, stream>>>(dst, dis, E3);
    k_rsqrt <<<cdiv(N3, T), T, 0, stream>>>(dis, N3);
    const int tiles = cdiv(N3, 16) * 1;
    k_gcn_gemm<32, 3><<<cdiv(tiles, WPB), T, 0, stream>>>(accA, 32, up3, dis, W4, g, accB, N3);
    k_scatter3<<<cdiv(E3, T), T, 0, stream>>>(src, dst, g, accB, E3);
    k_finalize<3><<<cdiv((long)N3 * 3, T), T, 0, stream>>>(accB, dis, b4, (long)N3 * 3);
  }
  // ---------------- final unpool to output ----------------
  k_gather3<<<cdiv(N4, T), T, 0, stream>>>(accB, up4, (float*)d_out, N4);
}
